// DenseEncoderLayer_27642409517358
// MI455X (gfx1250) — matrix operand order repaired and checked by reading, not compiled, not run on hardware
//
#include <hip/hip_runtime.h>
#include <hip/hip_bf16.h>

typedef _Float16 h16;
typedef _Float16 v16h __attribute__((ext_vector_type(16)));
typedef _Float16 v8h  __attribute__((ext_vector_type(8)));
typedef float    v8f  __attribute__((ext_vector_type(8)));

#define B_      4
#define S_      1024
#define DIM_    1024
#define HEADS_  16
#define DH_     64
#define E_      8
#define HID_    4096
#define NTOK    (B_ * S_)
#define EPS_    1e-5f
#define SCALE_  0.125f  /* 64^-0.5 */

// ---------------- WMMA fragment helpers (wave32, 16x16x32 f16) ----------------
// A-frag 16x32: lane row = lane&15, half = lane>>4; elem t = K 16*half+t
// (one contiguous 32B load).
__device__ __forceinline__ v16h load_a_frag(const h16* base, int ld) {
    const int lane = threadIdx.x & 31;
    const int row  = lane & 15;
    const int hf   = lane >> 4;
    return *(const v16h*)(base + (long)row * ld + hf * 16);
}

// B-frag 32x16 from BT[N,K] row-major: lane col = lane&15, half = lane>>4;
// elem t = B[16*half+t, col] = BT[col, 16*half+t] -> one contiguous 32B load.
__device__ __forceinline__ v16h load_b_frag(const h16* base, int ld) {
    const int lane = threadIdx.x & 31;
    const int col  = lane & 15;
    const int hf   = lane >> 4;
    return *(const v16h*)(base + (long)col * ld + hf * 16);
}

__device__ __forceinline__ v8f wmma_f16(v16h a, v16h b, v8f c) {
    return __builtin_amdgcn_wmma_f32_16x16x32_f16(false, a, false, b, (short)0, c,
                                                  false, false);
}

// Each wave computes a 16(M) x 64(N) strip: 4 C tiles, A-frag reused 4x.
__device__ __forceinline__ void gemm_wave_16x64(const h16* A, const h16* BT, int K,
                                                long m0, long n0, v8f c[4]) {
    for (int kb = 0; kb < K; kb += 32) {
        v16h a = load_a_frag(A + m0 * K + kb, K);
#pragma unroll
        for (int j = 0; j < 4; j++) {
            v16h b = load_b_frag(BT + (n0 + 16 * j) * (long)K + kb, K);
            c[j] = wmma_f16(a, b, c[j]);
        }
    }
}

// Block = 256 threads = 8 waves arranged 4(M) x 2(N): block tile 64 x 128.
__device__ __forceinline__ void wave_tile(long& m0, long& n0) {
    int w = threadIdx.x >> 5;
    m0 = (long)blockIdx.y * 64 + (long)(w >> 1) * 16;
    n0 = (long)blockIdx.x * 128 + (long)(w & 1) * 64;
}

// ---------------- GEMM kernels with varying epilogues ----------------
__global__ void k_gemm_f32(const h16* A, const h16* BT, float* C, int N, int K) {
    long m0, n0; wave_tile(m0, n0);
    v8f c[4];
#pragma unroll
    for (int j = 0; j < 4; j++)
#pragma unroll
        for (int r = 0; r < 8; r++) c[j][r] = 0.f;
    gemm_wave_16x64(A, BT, K, m0, n0, c);
    const int lane = threadIdx.x & 31, col = lane & 15, hf = lane >> 4;
#pragma unroll
    for (int j = 0; j < 4; j++)
#pragma unroll
        for (int r = 0; r < 8; r++)
            C[(m0 + 8 * hf + r) * (long)N + n0 + 16 * j + col] = c[j][r];
}

__global__ void k_gemm_f16(const h16* A, const h16* BT, h16* C, int N, int K) {
    long m0, n0; wave_tile(m0, n0);
    v8f c[4];
#pragma unroll
    for (int j = 0; j < 4; j++)
#pragma unroll
        for (int r = 0; r < 8; r++) c[j][r] = 0.f;
    gemm_wave_16x64(A, BT, K, m0, n0, c);
    const int lane = threadIdx.x & 31, col = lane & 15, hf = lane >> 4;
#pragma unroll
    for (int j = 0; j < 4; j++)
#pragma unroll
        for (int r = 0; r < 8; r++)
            C[(m0 + 8 * hf + r) * (long)N + n0 + 16 * j + col] = (h16)c[j][r];
}

// MoE up-projection: bias + tanh-approx GELU, f16 output.
__global__ void k_gemm_up(const h16* A, const h16* BT, const float* b1e, h16* H,
                          int N, int K) {
    long m0, n0; wave_tile(m0, n0);
    v8f c[4];
#pragma unroll
    for (int j = 0; j < 4; j++)
#pragma unroll
        for (int r = 0; r < 8; r++) c[j][r] = 0.f;
    gemm_wave_16x64(A, BT, K, m0, n0, c);
    const int lane = threadIdx.x & 31, col = lane & 15, hf = lane >> 4;
#pragma unroll
    for (int j = 0; j < 4; j++) {
        int cn = (int)n0 + 16 * j + col;
        float bias = b1e[cn];
#pragma unroll
        for (int r = 0; r < 8; r++) {
            float xv = c[j][r] + bias;
            float t  = tanhf(0.7978845608028654f * (xv + 0.044715f * xv * xv * xv));
            H[(m0 + 8 * hf + r) * (long)N + cn] = (h16)(0.5f * xv * (1.0f + t));
        }
    }
}

// MoE down-projection: bias + per-row gate weight, accumulate into out.
__global__ void k_gemm_down(const h16* A, const h16* BT, const float* b2e,
                            const float* gates, int e, int first, float* Out,
                            int N, int K) {
    long m0, n0; wave_tile(m0, n0);
    v8f c[4];
#pragma unroll
    for (int j = 0; j < 4; j++)
#pragma unroll
        for (int r = 0; r < 8; r++) c[j][r] = 0.f;
    gemm_wave_16x64(A, BT, K, m0, n0, c);
    const int lane = threadIdx.x & 31, col = lane & 15, hf = lane >> 4;
#pragma unroll
    for (int r = 0; r < 8; r++) {
        long row = m0 + 8 * hf + r;
        float g = gates[row * E_ + e];
#pragma unroll
        for (int j = 0; j < 4; j++) {
            int cn = (int)n0 + 16 * j + col;
            float v = g * (c[j][r] + b2e[cn]);
            if (first) Out[row * (long)N + cn] = v;
            else       Out[row * (long)N + cn] += v;
        }
    }
}

// ---------------- data movement / prep kernels ----------------
__global__ void k_f32_to_f16(const float* in, h16* out, long n) {
    long i = (long)blockIdx.x * blockDim.x + threadIdx.x;
    if (i < n) out[i] = (h16)in[i];
}

// in [R,C] f32 row-major  ->  out [C,R] f16 row-major (batched via blockIdx.z)
__global__ void k_transpose_f16(const float* in, h16* out, int R, int C) {
    long base = (long)blockIdx.z * R * C;
    long i = (long)blockIdx.x * blockDim.x + threadIdx.x;
    if (i < (long)R * C) {
        int r = (int)(i / C), c = (int)(i % C);
        out[base + (long)c * R + r] = (h16)in[base + i];
    }
}

// Per (b,h,s) row: qk-LayerNorm over DH=64 -> qn/kn f16 [b,h,s,64];
// V converted + transposed -> vT f16 [b,h,64,s]. One wave per row; lane owns
// dh = lane and lane+32; mean/var via full-wave shfl_xor reductions.
__global__ void k_prep_qkv(const float* qkv, const float* qg, const float* kg,
                           h16* qn, h16* kn, h16* vT) {
    int gid  = blockIdx.x * blockDim.x + threadIdx.x;
    int wv   = gid >> 5;            // 0 .. B*H*S-1
    int lane = threadIdx.x & 31;
    int s  = wv & (S_ - 1);
    int bh = wv >> 10;
    int h  = bh & (HEADS_ - 1);
    int b  = bh >> 4;
    long tok = (long)b * S_ + s;
    const float* rowq = qkv + tok * (3 * DIM_) + h * DH_;
    const float* rowk = rowq + DIM_;
    const float* rowv = rowq + 2 * DIM_;
    long obase = ((long)bh * S_ + s) * DH_;

    // Q layernorm
    {
        float a0 = rowq[lane], a1 = rowq[lane + 32];
        float sum = a0 + a1;
#pragma unroll
        for (int m = 16; m >= 1; m >>= 1) sum += __shfl_xor(sum, m, 32);
        float mu = sum * (1.0f / 64.0f);
        float d0 = a0 - mu, d1 = a1 - mu;
        float vs = d0 * d0 + d1 * d1;
#pragma unroll
        for (int m = 16; m >= 1; m >>= 1) vs += __shfl_xor(vs, m, 32);
        float rstd = rsqrtf(vs * (1.0f / 64.0f) + EPS_);
        qn[obase + lane]      = (h16)(d0 * rstd * qg[lane]);
        qn[obase + lane + 32] = (h16)(d1 * rstd * qg[lane + 32]);
    }
    // K layernorm
    {
        float a0 = rowk[lane], a1 = rowk[lane + 32];
        float sum = a0 + a1;
#pragma unroll
        for (int m = 16; m >= 1; m >>= 1) sum += __shfl_xor(sum, m, 32);
        float mu = sum * (1.0f / 64.0f);
        float d0 = a0 - mu, d1 = a1 - mu;
        float vs = d0 * d0 + d1 * d1;
#pragma unroll
        for (int m = 16; m >= 1; m >>= 1) vs += __shfl_xor(vs, m, 32);
        float rstd = rsqrtf(vs * (1.0f / 64.0f) + EPS_);
        kn[obase + lane]      = (h16)(d0 * rstd * kg[lane]);
        kn[obase + lane + 32] = (h16)(d1 * rstd * kg[lane + 32]);
    }
    // V transpose
    vT[((long)bh * DH_ + lane) * S_ + s]        = (h16)rowv[lane];
    vT[((long)bh * DH_ + lane + 32) * S_ + s]   = (h16)rowv[lane + 32];
}

// Flash attention: one wave (32 threads) per (bh, 16-query block).
__global__ void k_attn(const h16* qn, const h16* kn, const h16* vT, h16* ao) {
    __shared__ h16 sP[16 * 32];
    const int bh = blockIdx.y;
    const int h  = bh & (HEADS_ - 1);
    const int b  = bh >> 4;
    const int qbase = blockIdx.x * 16;
    const int lane = threadIdx.x & 31;
    const int col = lane & 15, hf = lane >> 4;

    const h16* qbh = qn + (long)bh * S_ * DH_;
    const h16* kbh = kn + (long)bh * S_ * DH_;
    const h16* vbh = vT + (long)bh * DH_ * S_;

    v16h aq0 = load_a_frag(qbh + (long)qbase * DH_,      DH_);  // dh 0..31
    v16h aq1 = load_a_frag(qbh + (long)qbase * DH_ + 32, DH_);  // dh 32..63

    v8f co[4];
    float m[8], l[8];
#pragma unroll
    for (int j = 0; j < 4; j++)
#pragma unroll
        for (int r = 0; r < 8; r++) co[j][r] = 0.f;
#pragma unroll
    for (int r = 0; r < 8; r++) { m[r] = -1e30f; l[r] = 0.f; }

    for (int kb = 0; kb <= qbase; kb += 32) {
        // scores: 2 tiles of 16q x 16k, each from 2 WMMAs over dh
        v8f sc[2];
#pragma unroll
        for (int t = 0; t < 2; t++) {
            v16h bk0 = load_b_frag(kbh + (long)(kb + 16 * t) * DH_,      DH_);
            v16h bk1 = load_b_frag(kbh + (long)(kb + 16 * t) * DH_ + 32, DH_);
            v8f s;
#pragma unroll
            for (int r = 0; r < 8; r++) s[r] = 0.f;
            s = wmma_f16(aq0, bk0, s);
            s = wmma_f16(aq1, bk1, s);
            sc[t] = s;
        }
        // scale + causal mask + per-row max
        float vmax[8];
#pragma unroll
        for (int r = 0; r < 8; r++) {
            int qi = qbase + 8 * hf + r;
#pragma unroll
            for (int t = 0; t < 2; t++) {
                int ki = kb + 16 * t + col;
                float v = sc[t][r] * SCALE_;
                if (ki > qi) v = -1e30f;
                sc[t][r] = v;
            }
            vmax[r] = fmaxf(sc[0][r], sc[1][r]);
        }
#pragma unroll
        for (int r = 0; r < 8; r++)
#pragma unroll
            for (int msk = 1; msk < 16; msk <<= 1)
                vmax[r] = fmaxf(vmax[r], __shfl_xor(vmax[r], msk, 32));
        // online softmax update
        float alpha[8];
#pragma unroll
        for (int r = 0; r < 8; r++) {
            float nm = fmaxf(m[r], vmax[r]);
            alpha[r] = __expf(m[r] - nm);
            m[r] = nm;
            float p0 = __expf(sc[0][r] - nm);
            float p1 = __expf(sc[1][r] - nm);
            sc[0][r] = p0; sc[1][r] = p1;
            float rs = p0 + p1;
#pragma unroll
            for (int msk = 1; msk < 16; msk <<= 1)
                rs += __shfl_xor(rs, msk, 32);
            l[r] = l[r] * alpha[r] + rs;
        }
#pragma unroll
        for (int j = 0; j < 4; j++)
#pragma unroll
            for (int r = 0; r < 8; r++) co[j][r] *= alpha[r];
        // P: C-layout -> A-layout via LDS
#pragma unroll
        for (int r = 0; r < 8; r++) {
            sP[(8 * hf + r) * 32 + col]      = (h16)sc[0][r];
            sP[(8 * hf + r) * 32 + 16 + col] = (h16)sc[1][r];
        }
        __syncthreads();
        v16h pa = load_a_frag(sP, 32);
        __syncthreads();
        // O += P @ V  (4 dh groups of 16)
#pragma unroll
        for (int j = 0; j < 4; j++) {
            v16h bv = load_b_frag(vbh + (long)(16 * j) * S_ + kb, S_);
            co[j] = wmma_f16(pa, bv, co[j]);
        }
    }
    // normalize and store [b, s, h*64 + dh] as f16
    long tokbase = (long)b * S_ + qbase;
#pragma unroll
    for (int j = 0; j < 4; j++)
#pragma unroll
        for (int r = 0; r < 8; r++) {
            long tok = tokbase + 8 * hf + r;
            ao[tok * DIM_ + h * DH_ + 16 * j + col] = (h16)(co[j][r] / l[r]);
        }
}

// Gate: o @ gate_w + gate_b, softmax over E=8. One token per thread.
__global__ void k_gate(const h16* o16, const float* gw, const float* gb, float* gates) {
    int t = blockIdx.x * blockDim.x + threadIdx.x;
    if (t >= NTOK) return;
    float acc[E_];
#pragma unroll
    for (int e = 0; e < E_; e++) acc[e] = gb[e];
    const h16* row = o16 + (long)t * DIM_;
    for (int k = 0; k < DIM_; k++) {
        float xv = (float)row[k];
        const float* g = gw + k * E_;
#pragma unroll
        for (int e = 0; e < E_; e++) acc[e] += xv * g[e];
    }
    float mx = acc[0];
#pragma unroll
    for (int e = 1; e < E_; e++) mx = fmaxf(mx, acc[e]);
    float ssum = 0.f;
#pragma unroll
    for (int e = 0; e < E_; e++) { acc[e] = __expf(acc[e] - mx); ssum += acc[e]; }
    float inv = 1.f / ssum;
#pragma unroll
    for (int e = 0; e < E_; e++) gates[(long)t * E_ + e] = acc[e] * inv;
}

// ---------------- host launch ----------------
extern "C" void kernel_launch(void* const* d_in, const int* in_sizes, int n_in,
                              void* d_out, int out_size, void* d_ws, size_t ws_size,
                              hipStream_t stream) {
    const float* x    = (const float*)d_in[0];
    const float* wqkv = (const float*)d_in[1];
    const float* wout = (const float*)d_in[2];
    const float* qg   = (const float*)d_in[3];
    const float* kg   = (const float*)d_in[4];
    const float* gw   = (const float*)d_in[5];
    const float* gb   = (const float*)d_in[6];
    const float* w1   = (const float*)d_in[7];
    const float* b1   = (const float*)d_in[8];
    const float* w2   = (const float*)d_in[9];
    const float* b2   = (const float*)d_in[10];
    float* out = (float*)d_out;

    char* p = (char*)d_ws;
    auto alloc = [&](size_t bytes) -> char* {
        char* q = p; p += (bytes + 255) & ~(size_t)255; return q;
    };
    h16*   x16   = (h16*)  alloc((size_t)NTOK * DIM_ * 2);
    h16*   wqkvT = (h16*)  alloc((size_t)3 * DIM_ * DIM_ * 2);
    h16*   woutT = (h16*)  alloc((size_t)DIM_ * DIM_ * 2);
    h16*   w1T   = (h16*)  alloc((size_t)E_ * HID_ * DIM_ * 2);
    h16*   w2T   = (h16*)  alloc((size_t)E_ * DIM_ * HID_ * 2);
    float* qkv   = (float*)alloc((size_t)NTOK * 3 * DIM_ * 4);
    h16*   qn    = (h16*)  alloc((size_t)NTOK * DIM_ * 2);
    h16*   kn    = (h16*)  alloc((size_t)NTOK * DIM_ * 2);
    h16*   vT    = (h16*)  alloc((size_t)NTOK * DIM_ * 2);
    h16*   ao    = (h16*)  alloc((size_t)NTOK * DIM_ * 2);
    h16*   o16   = (h16*)  alloc((size_t)NTOK * DIM_ * 2);
    float* gates = (float*)alloc((size_t)NTOK * E_ * 4);
    h16*   hbuf  = (h16*)  alloc((size_t)NTOK * HID_ * 2);

    dim3 blk(256);
    // precision conversion / weight transposition
    k_f32_to_f16<<<(NTOK * DIM_ + 255) / 256, blk, 0, stream>>>(x, x16, (long)NTOK * DIM_);
    k_transpose_f16<<<dim3((DIM_ * 3 * DIM_ + 255) / 256, 1, 1), blk, 0, stream>>>(
        wqkv, wqkvT, DIM_, 3 * DIM_);
    k_transpose_f16<<<dim3((DIM_ * DIM_ + 255) / 256, 1, 1), blk, 0, stream>>>(
        wout, woutT, DIM_, DIM_);
    k_transpose_f16<<<dim3((DIM_ * HID_ + 255) / 256, 1, E_), blk, 0, stream>>>(
        w1, w1T, DIM_, HID_);
    k_transpose_f16<<<dim3((HID_ * DIM_ + 255) / 256, 1, E_), blk, 0, stream>>>(
        w2, w2T, HID_, DIM_);
    // qkv = x @ w_qkv   [4096 x 3072], K=1024
    k_gemm_f32<<<dim3(3 * DIM_ / 128, NTOK / 64), blk, 0, stream>>>(
        x16, wqkvT, qkv, 3 * DIM_, DIM_);
    // qk-norm + v transpose
    k_prep_qkv<<<(B_ * HEADS_ * S_) / 8, blk, 0, stream>>>(qkv, qg, kg, qn, kn, vT);
    // causal flash attention
    k_attn<<<dim3(S_ / 16, B_ * HEADS_), dim3(32), 0, stream>>>(qn, kn, vT, ao);
    // out projection
    k_gemm_f16<<<dim3(DIM_ / 128, NTOK / 64), blk, 0, stream>>>(
        ao, woutT, o16, DIM_, DIM_);
    // gate softmax
    k_gate<<<(NTOK + 255) / 256, blk, 0, stream>>>(o16, gw, gb, gates);
    // dense MoE: per expert up (GELU) then gated down-accumulate into out
    for (int e = 0; e < E_; e++) {
        k_gemm_up<<<dim3(HID_ / 128, NTOK / 64), blk, 0, stream>>>(
            o16, w1T + (size_t)e * HID_ * DIM_, b1 + (size_t)e * HID_, hbuf, HID_, DIM_);
        k_gemm_down<<<dim3(DIM_ / 128, NTOK / 64), blk, 0, stream>>>(
            hbuf, w2T + (size_t)e * DIM_ * HID_, b2 + (size_t)e * DIM_, gates, e,
            (e == 0) ? 1 : 0, out, DIM_, HID_);
    }
}
